// h_diag_Land_42932493090900
// MI455X (gfx1250) — matrix operand order repaired and checked by reading, not compiled, not run on hardware
//
#include <hip/hip_runtime.h>
#include <stdint.h>

constexpr int Nsz = 8192;   // reference rows
constexpr int Dsz = 256;    // feature dim

typedef __bf16 v8bf  __attribute__((ext_vector_type(8)));
typedef __bf16 v16bf __attribute__((ext_vector_type(16)));
typedef float  v8f   __attribute__((ext_vector_type(8)));
typedef float  v4f   __attribute__((ext_vector_type(4)));

// ---- LDS staging geometry (padded to kill ds bank conflicts) ----
constexpr uint32_t REF_STRIDE  = 528;                              // 32 rows x 512B data, stride 528B (33*16)
constexpr uint32_t REFT_STRIDE = 80;                               // 256 rows x 64B data, stride 80B (5*16)
constexpr uint32_t OFF_REFT    = 32 * REF_STRIDE;                  // 16896
constexpr uint32_t OFF_REF2T   = OFF_REFT + 256 * REFT_STRIDE;     // 37376
constexpr uint32_t BUF_BYTES   = OFF_REF2T + 256 * REFT_STRIDE;    // 57856 per buffer
constexpr uint32_t OFF_WLDS    = 2 * BUF_BYTES;                    // 115712
constexpr uint32_t W_STRIDE    = 40;                               // bf16 elems per w-bounce row (80B)
constexpr uint32_t SMEM_BYTES  = OFF_WLDS + 4 * 16 * W_STRIDE * 2; // 120832

static __device__ __forceinline__ v16bf cat16(v8bf lo, v8bf hi) {
  return __builtin_shufflevector(lo, hi, 0,1,2,3,4,5,6,7,8,9,10,11,12,13,14,15);
}

static __device__ __forceinline__ v16bf ldfrag(const __bf16* p) {
  v8bf lo = *(const v8bf*)p;
  v8bf hi = *(const v8bf*)(p + 8);
  return cat16(lo, hi);
}

// 16B-per-lane async global->LDS copy (ASYNCcnt-tracked, CDNA5)
static __device__ __forceinline__ void async_cp16(uint32_t lds_off, const void* g) {
  asm volatile("global_load_async_to_lds_b128 %0, %1, off"
               :: "v"(lds_off), "v"((uint64_t)(uintptr_t)g) : "memory");
}

template <int NCNT>
static __device__ __forceinline__ void wait_async() {
#if __has_builtin(__builtin_amdgcn_s_wait_asynccnt)
  __builtin_amdgcn_s_wait_asynccnt(NCNT);
#else
  asm volatile("s_wait_asynccnt %0" :: "i"(NCNT) : "memory");
#endif
}

static __device__ __forceinline__ void block_sync() {
  asm volatile("" ::: "memory");
  __builtin_amdgcn_s_barrier();
  asm volatile("" ::: "memory");
}

// ---------------------------------------------------------------------------
// Prep: r_sq[n]; ref_bf16[N][D] row-major; chunk-contiguous transposed tiles
//   refTc / ref2Tc : per 32-row chunk c, a contiguous [256 d][32 n] bf16 block.
// ---------------------------------------------------------------------------
__global__ __launch_bounds__(256) void prep_kernel(
    const float* __restrict__ ref, float* __restrict__ rsq,
    __bf16* __restrict__ refb, __bf16* __restrict__ refTc,
    __bf16* __restrict__ ref2Tc) {
  __shared__ float tile[16 * Dsz];
  const int nb16 = blockIdx.x * 16;
  const int c    = nb16 >> 5;          // 32-row chunk index
  const int half = (nb16 >> 4) & 1;    // which 16-row half of the chunk
  const int t    = threadIdx.x;        // d index 0..255
  #pragma unroll
  for (int i = 0; i < 16; ++i) {
    float v = ref[(size_t)(nb16 + i) * Dsz + t];
    tile[i * Dsz + t] = v;
    refb[(size_t)(nb16 + i) * Dsz + t] = (__bf16)v;
  }
  __syncthreads();
  v8bf a0, a1, b0, b1;
  #pragma unroll
  for (int i = 0; i < 8; ++i) {
    float v = tile[i * Dsz + t];
    float w = tile[(i + 8) * Dsz + t];
    a0[i] = (__bf16)v;        a1[i] = (__bf16)w;
    b0[i] = (__bf16)(v * v);  b1[i] = (__bf16)(w * w);
  }
  __bf16* d1 = refTc  + (size_t)c * (256 * 32) + t * 32 + half * 16;
  __bf16* d2 = ref2Tc + (size_t)c * (256 * 32) + t * 32 + half * 16;
  *(v8bf*)(d1)     = a0;  *(v8bf*)(d1 + 8) = a1;
  *(v8bf*)(d2)     = b0;  *(v8bf*)(d2 + 8) = b1;
  if (t < 16) {
    float s = 0.f;
    #pragma unroll 8
    for (int d = 0; d < Dsz; ++d) { float v = tile[t * Dsz + d]; s += v * v; }
    rsq[nb16 + t] = s;
  }
}

// ---------------------------------------------------------------------------
// Stage one 32-row chunk (ref row-major + refT + ref2T, 48KB) into LDS.
// 24 async b128 instructions per wave -> ASYNCcnt += 24 per wave.
// ---------------------------------------------------------------------------
static __device__ __forceinline__ void stage_chunk(
    const __bf16* refb, const __bf16* refTc, const __bf16* ref2Tc,
    int nb, uint32_t lbase, int tid) {
  const int c = nb >> 5;
  {  // region 0: ref rows, 32 x 512B -> padded stride 528B
    const char* src = (const char*)(refb + (size_t)nb * Dsz);
    const int row = tid >> 5;             // 0..3
    const int col = (tid & 31) * 16;      // 0..496
    #pragma unroll
    for (int r = 0; r < 8; ++r) {
      const int gr = row + r * 4;         // 0..31
      async_cp16(lbase + gr * REF_STRIDE + col, src + gr * 512 + col);
    }
  }
  {  // region 1: refT chunk, 256 x 64B -> padded stride 80B
    const char* src = (const char*)(refTc + (size_t)c * (256 * 32));
    const int row = tid >> 2;             // 0..31
    const int col = (tid & 3) * 16;       // 0..48
    #pragma unroll
    for (int r = 0; r < 8; ++r) {
      const int gr = row + r * 32;        // 0..255
      async_cp16(lbase + OFF_REFT + gr * REFT_STRIDE + col, src + gr * 64 + col);
    }
  }
  {  // region 2: ref2T chunk
    const char* src = (const char*)(ref2Tc + (size_t)c * (256 * 32));
    const int row = tid >> 2;
    const int col = (tid & 3) * 16;
    #pragma unroll
    for (int r = 0; r < 8; ++r) {
      const int gr = row + r * 32;
      async_cp16(lbase + OFF_REF2T + gr * REFT_STRIDE + col, src + gr * 64 + col);
    }
  }
}

// ---------------------------------------------------------------------------
// Main fused kernel: 4 waves/block, each wave owns 16 rows of x; the block
// cooperatively double-buffers ref chunks in LDS via async copies.
// ---------------------------------------------------------------------------
__global__ __launch_bounds__(128, 1) void hdiag_kernel(
    const float* __restrict__ x, const float* __restrict__ Wp,
    const float* __restrict__ rsq, const __bf16* __restrict__ refb,
    const __bf16* __restrict__ refTc, const __bf16* __restrict__ ref2Tc,
    float* __restrict__ out) {
  extern __shared__ char smem[];
  const int tid  = threadIdx.x;
  const int lane = tid & 31;
  const int wave = tid >> 5;
  const int rowbase = blockIdx.x * 64 + wave * 16;
  const int lmod = lane & 15;
  const int lhi  = lane >> 4;
  const uint32_t lds0 = (uint32_t)(uintptr_t)(void*)smem;   // wave-relative LDS base
  __bf16* wlds = (__bf16*)(smem + OFF_WLDS + wave * (16 * W_STRIDE * 2));

  // ---- x_sq for this wave's 16 rows, broadcast into C-tile layout ----
  float part = 0.f;
  {
    const float* xr = x + (size_t)(rowbase + lmod) * Dsz + lhi * 128;
    #pragma unroll 4
    for (int d = 0; d < 128; d += 4) {
      v4f v = *(const v4f*)(xr + d);
      part += v.x * v.x + v.y * v.y + v.z * v.z + v.w * v.w;
    }
  }
  float xsqrow = part + __shfl_xor(part, 16, 32);   // lane L holds xsq[L%16]
  float xsq_c[8];
  #pragma unroll
  for (int j = 0; j < 8; ++j) xsq_c[j] = __shfl(xsqrow, j + 8 * lhi, 32);

  // ---- resident bf16 A-fragments of the x tile (K = D, 8 steps of 32) ----
  v16bf axf[8];
  #pragma unroll
  for (int ks = 0; ks < 8; ++ks) {
    const float* p0 = x + (size_t)(rowbase + lmod) * Dsz + ks * 32 + 16 * lhi;
    v16bf f;
    #pragma unroll
    for (int e = 0; e < 8; ++e) {
      f[e]     = (__bf16)p0[e];
      f[e + 8] = (__bf16)p0[e + 8];
    }
    axf[ks] = f;
  }

  v8f acc1[16], acc2[16];
  #pragma unroll
  for (int t = 0; t < 16; ++t) { acc1[t] = (v8f){}; acc2[t] = (v8f){}; }
  float wsum_p[8] = {0.f, 0.f, 0.f, 0.f, 0.f, 0.f, 0.f, 0.f};
  const float coef = -0.011270898756944714f;  // -log2(e) / (2*gamma^2)

  // ---- prime the double buffer ----
  stage_chunk(refb, refTc, ref2Tc, 0,  lds0,             tid);
  stage_chunk(refb, refTc, ref2Tc, 32, lds0 + BUF_BYTES, tid);

  for (int nb = 0; nb < Nsz; nb += 32) {
    const uint32_t bufbase = ((nb >> 5) & 1) ? BUF_BYTES : 0u;
    // retire the chunk we are about to consume (next chunk stays in flight)
    if (nb + 32 < Nsz) wait_async<24>(); else wait_async<0>();
    block_sync();   // all waves' copies for this chunk are visible

    const char* bref = smem + bufbase;
    const char* bT1  = smem + bufbase + OFF_REFT;
    const char* bT2  = smem + bufbase + OFF_REF2T;

    const float rv0 = rsq[nb + lmod];
    const float rv1 = rsq[nb + 16 + lmod];

    // ---- cross GEMM: two 16x16 tiles, two-deep pipelined fragment loads ----
    const __bf16* bp0 = (const __bf16*)(bref + (size_t)lmod * REF_STRIDE) + 16 * lhi;
    const __bf16* bp1 =
        (const __bf16*)(bref + (size_t)(16 + lmod) * REF_STRIDE) + 16 * lhi;
    v8f cc0 = (v8f){}, cc1 = (v8f){};
    v16bf f0 = ldfrag(bp0);
    v16bf f1 = ldfrag(bp0 + 32);
    #pragma unroll
    for (int s = 0; s < 16; ++s) {
      v16bf fn = f0;   // dummy on the last two steps
      if (s < 14) {
        const int s2 = s + 2;
        fn = ldfrag(((s2 < 8) ? bp0 : bp1) + (s2 & 7) * 32);
      }
      if (s < 8)
        cc0 = __builtin_amdgcn_wmma_f32_16x16x32_bf16(
                  false, axf[s], false, f0, (short)0, cc0, false, false);
      else
        cc1 = __builtin_amdgcn_wmma_f32_16x16x32_bf16(
                  false, axf[s & 7], false, f0, (short)0, cc1, false, false);
      f0 = f1;
      f1 = fn;
    }

    // ---- weights + per-lane wsum partials + LDS bounce (C-layout -> A-layout)
    #pragma unroll
    for (int j = 0; j < 8; ++j) {
      float w0 = __builtin_amdgcn_exp2f((xsq_c[j] + rv0 - 2.0f * cc0[j]) * coef);
      float w1 = __builtin_amdgcn_exp2f((xsq_c[j] + rv1 - 2.0f * cc1[j]) * coef);
      wsum_p[j] += w0 + w1;
      wlds[(j + 8 * lhi) * W_STRIDE + lmod]      = (__bf16)w0;
      wlds[(j + 8 * lhi) * W_STRIDE + 16 + lmod] = (__bf16)w1;
    }
    // issue first two stage-2 B fragments early (independent of w)
    v16bf g0 = ldfrag((const __bf16*)(bT1 + (size_t)lmod * REFT_STRIDE) + 16 * lhi);
    v16bf g1 = ldfrag((const __bf16*)(bT2 + (size_t)lmod * REFT_STRIDE) + 16 * lhi);
    asm volatile("" ::: "memory");   // per-wave LDS is in-order; stop reordering
    v16bf aw = cat16(*(const v8bf*)(wlds + lmod * W_STRIDE + 16 * lhi),
                     *(const v8bf*)(wlds + lmod * W_STRIDE + 16 * lhi + 8));
    asm volatile("" ::: "memory");

    // ---- stage 2: flattened 32-step chain (b1(t), b2(t) alternating),
    //      two fragments always in flight ----
    #pragma unroll
    for (int u = 0; u < 32; ++u) {
      v16bf gn = g0;   // dummy on the last two steps
      if (u < 30) {
        const int u2 = u + 2;
        const char* base = (u2 & 1) ? bT2 : bT1;
        gn = ldfrag((const __bf16*)(base + (size_t)((u2 >> 1) * 16 + lmod) *
                                               REFT_STRIDE) + 16 * lhi);
      }
      const int t = u >> 1;
      if ((u & 1) == 0)
        acc1[t] = __builtin_amdgcn_wmma_f32_16x16x32_bf16(
                      false, aw, false, g0, (short)0, acc1[t], false, false);
      else
        acc2[t] = __builtin_amdgcn_wmma_f32_16x16x32_bf16(
                      false, aw, false, g0, (short)0, acc2[t], false, false);
      g0 = g1;
      g1 = gn;
    }

    block_sync();   // everyone done reading this buffer
    if (nb + 64 < Nsz)
      stage_chunk(refb, refTc, ref2Tc, nb + 64, lds0 + bufbase, tid);
  }

  // ---- wsum: reduce over the 16 lanes sharing each row (lands in C-layout) ----
  #pragma unroll
  for (int j = 0; j < 8; ++j) {
    float s = wsum_p[j];
    s += __shfl_xor(s, 1, 32);
    s += __shfl_xor(s, 2, 32);
    s += __shfl_xor(s, 4, 32);
    s += __shfl_xor(s, 8, 32);
    wsum_p[j] = s;
  }

  // ---- combine: M = acc2 - 2*x*acc1 + x^2*wsum ; out = 1/(W*M + eps) ----
  const float Wv = Wp[0];
  #pragma unroll
  for (int t = 0; t < 16; ++t) {
    #pragma unroll
    for (int j = 0; j < 8; ++j) {
      const int b = rowbase + j + 8 * lhi;
      const int d = t * 16 + lmod;
      const float xv = x[(size_t)b * Dsz + d];
      const float m = acc2[t][j] - 2.0f * xv * acc1[t][j] + xv * xv * wsum_p[j];
      out[(size_t)b * Dsz + d] = 1.0f / (Wv * m + 0.001f);
    }
  }
}

// ---------------------------------------------------------------------------
extern "C" void kernel_launch(void* const* d_in, const int* in_sizes, int n_in,
                              void* d_out, int out_size, void* d_ws, size_t ws_size,
                              hipStream_t stream) {
  const float* x   = (const float*)d_in[0];
  const float* ref = (const float*)d_in[1];
  const float* W   = (const float*)d_in[2];
  float* out = (float*)d_out;

  // workspace: r_sq | ref_bf16 | refTc | ref2Tc  (~12.03 MB)
  char* ws = (char*)d_ws;
  float*  rsq    = (float*)ws;
  __bf16* refb   = (__bf16*)(ws + 32768);
  __bf16* refTc  = (__bf16*)(ws + 32768 + (size_t)Nsz * Dsz * 2);
  __bf16* ref2Tc = (__bf16*)(ws + 32768 + (size_t)Nsz * Dsz * 4);

  prep_kernel<<<Nsz / 16, 256, 0, stream>>>(ref, rsq, refb, refTc, ref2Tc);
  hdiag_kernel<<<8192 / 64, 128, SMEM_BYTES, stream>>>(x, W, rsq, refb, refTc,
                                                       ref2Tc, out);
}
